// HierarchicalGNN_52879637348373
// MI455X (gfx1250) — matrix operand order repaired and checked by reading, not compiled, not run on hardware
//
#include <hip/hip_runtime.h>
#include <hip/hip_bf16.h>
#include <math.h>

// ---------------------------------------------------------------------------
// HierarchicalGNN on MI455X (gfx1250): f16 WMMA GEMMs (f32 accum, 16x64
// register-blocked per wave) + atomic segment ops for GAT attention/pooling.
// ---------------------------------------------------------------------------

typedef __attribute__((ext_vector_type(16))) _Float16 v16h;
typedef __attribute__((ext_vector_type(8)))  _Float16 v8h;
typedef __attribute__((ext_vector_type(8)))  float    v8f;

#define Nn   50000
#define Ee   800000
#define EP   (Ee + Nn)
#define FIN  128
#define Hh   4
#define Cc   64
#define Dd   256
#define Gg   64
#define NFn  2000
#define OUTd 128

// ------------------------------- helpers -----------------------------------

__global__ void fill_f32_kernel(float* p, float v, size_t n) {
  size_t i = (size_t)blockIdx.x * blockDim.x + threadIdx.x;
  size_t st = (size_t)gridDim.x * blockDim.x;
  for (; i < n; i += st) p[i] = v;
}

__global__ void cvt_f32_f16_kernel(const float* __restrict__ in,
                                   _Float16* __restrict__ out, size_t n) {
  size_t i = (size_t)blockIdx.x * blockDim.x + threadIdx.x;
  size_t st = (size_t)gridDim.x * blockDim.x;
  for (; i < n; i += st) out[i] = (_Float16)in[i];
}

// W[K,Nd] (row major f32) -> Wt[Nd,K] (row major f16)
__global__ void transpose_w_kernel(const float* __restrict__ W,
                                   _Float16* __restrict__ Wt, int K, int Nd) {
  size_t n = (size_t)K * Nd;
  size_t i = (size_t)blockIdx.x * blockDim.x + threadIdx.x;
  size_t st = (size_t)gridDim.x * blockDim.x;
  for (; i < n; i += st) {
    int k = (int)(i / Nd);
    int c = (int)(i % Nd);
    Wt[(size_t)c * K + k] = (_Float16)W[i];
  }
}

__device__ __forceinline__ void atomicMaxF(float* addr, float val) {
  unsigned* ua = reinterpret_cast<unsigned*>(addr);
  unsigned old = __float_as_uint(*addr);
  while (__uint_as_float(old) < val) {
    unsigned assumed = old;
    old = atomicCAS(ua, assumed, __float_as_uint(val));
    if (old == assumed) break;
  }
}

__device__ __forceinline__ int edge_src(const int* ei, int idx) {
  return (idx < Ee) ? ei[idx] : (idx - Ee);
}
__device__ __forceinline__ int edge_dst(const int* ei, int idx) {
  return (idx < Ee) ? ei[Ee + idx] : (idx - Ee);
}

// ------------------------------ WMMA GEMM ----------------------------------
// C[M,Nc](f32) = A[M,K](f16, row-major) * Bt[Nc,K](f16, row-major)^T (+bias,act)
// Register-blocked: each wave computes a 16(M) x 64(N) strip using one shared
// A fragment and 4 B fragments per 32-deep K step (4 WMMAs / 10 b128 loads).
// 4 waves per block cover 256 output columns. K mult of 32, M mult of 16,
// Nc mult of 64. act: 0=none, 1=elu.
__global__ void gemm_wmma_kernel(const _Float16* __restrict__ A,
                                 const _Float16* __restrict__ Bt,
                                 const float* __restrict__ bias,
                                 float* __restrict__ C,
                                 int M, int K, int Nc, int act) {
  const int lane   = threadIdx.x & 31;
  const int wave   = threadIdx.x >> 5;
  const int tileM  = blockIdx.x;
  const int tileN0 = (blockIdx.y * 4 + wave) * 64;     // first of 4 N-tiles
  if (tileM * 16 >= M || tileN0 >= Nc) return;         // wave-uniform

  const int row = lane & 15;   // A: M row / B: N col / C: N col
  const int hi  = lane >> 4;   // K-half selector

  // A lane base: row (tileM*16+row), first element at K = hi*8
  const _Float16* Arow = A + (size_t)(tileM * 16 + row) * K + hi * 8;
  // B lane bases: col (tileN0 + j*16 + row), first element at K = hi*8
  const _Float16* Brow[4];
#pragma unroll
  for (int j = 0; j < 4; ++j)
    Brow[j] = Bt + (size_t)(tileN0 + j * 16 + row) * K + hi * 8;

  union AV { v16h v; v8h h[2]; };
  v8f acc[4] = {v8f{}, v8f{}, v8f{}, v8f{}};

  for (int k0 = 0; k0 < K; k0 += 32) {
    __builtin_prefetch(Arow + k0 + 64, 0, 1);   // -> global_prefetch_b8
    AV a;
    a.h[0] = *(const v8h*)(Arow + k0);          // K = k0 + hi*8 + [0..7]
    a.h[1] = *(const v8h*)(Arow + k0 + 16);     // K = k0 + 16 + hi*8 + [0..7]
    AV b[4];
#pragma unroll
    for (int j = 0; j < 4; ++j) {
      b[j].h[0] = *(const v8h*)(Brow[j] + k0);      // K = k0 + hi*8 + [0..7]
      b[j].h[1] = *(const v8h*)(Brow[j] + k0 + 16); // K = k0 + 16 + hi*8 + [0..7]
    }
#pragma unroll
    for (int j = 0; j < 4; ++j)
      acc[j] = __builtin_amdgcn_wmma_f32_16x16x32_f16(
          /*neg_a=*/false, a.v, /*neg_b=*/false, b[j].v,
          /*c_mod=*/(short)0, acc[j], /*reuse_a=*/false, /*reuse_b=*/false);
  }

#pragma unroll
  for (int j = 0; j < 4; ++j) {
    const int col = tileN0 + j * 16 + row;
    const float bv = bias ? bias[col] : 0.0f;
#pragma unroll
    for (int i = 0; i < 8; ++i) {
      int Mr = tileM * 16 + i + hi * 8;    // C: VGPR i -> M = i (+8 for hi lanes)
      float v = acc[j][i] + bv;
      if (act == 1) v = (v > 0.0f) ? v : (__expf(v) - 1.0f);
      C[(size_t)Mr * Nc + col] = v;
    }
  }
}

// ---------------------------- GAT attention --------------------------------

// s_src[n,h] = sum_c h[n,h*C+c]*a_src[h,c]  (and same for dst)
__global__ void attn_scores_kernel(const float* __restrict__ h,
                                   const float* __restrict__ a_src,
                                   const float* __restrict__ a_dst,
                                   float* __restrict__ s_src,
                                   float* __restrict__ s_dst) {
  int i = blockIdx.x * blockDim.x + threadIdx.x;   // over Nn*Hh
  if (i >= Nn * Hh) return;
  int node = i / Hh, hd = i % Hh;
  const float* hp = h + (size_t)node * Dd + hd * Cc;
  const float* as = a_src + hd * Cc;
  const float* ad = a_dst + hd * Cc;
  float ss = 0.0f, sd = 0.0f;
#pragma unroll 4
  for (int c = 0; c < Cc; ++c) { float v = hp[c]; ss += v * as[c]; sd += v * ad[c]; }
  s_src[i] = ss; s_dst[i] = sd;
}

// e = leaky_relu(s_src[src]+s_dst[dst]); segment-max into m[dst]
__global__ void edge_logits_kernel(const int* __restrict__ ei,
                                   const float* __restrict__ s_src,
                                   const float* __restrict__ s_dst,
                                   float* __restrict__ ebuf,
                                   float* __restrict__ m) {
  int e = blockIdx.x * blockDim.x + threadIdx.x;
  if (e >= EP) return;
  int s = edge_src(ei, e), d = edge_dst(ei, e);
#pragma unroll
  for (int hd = 0; hd < Hh; ++hd) {
    float v = s_src[s * Hh + hd] + s_dst[d * Hh + hd];
    v = (v > 0.0f) ? v : 0.2f * v;
    ebuf[(size_t)e * Hh + hd] = v;
    atomicMaxF(&m[d * Hh + hd], v);
  }
}

// ex = exp(e - m[dst]); segment-sum into ssum[dst]
__global__ void edge_exp_kernel(const int* __restrict__ ei,
                                const float* __restrict__ m,
                                float* __restrict__ ebuf,
                                float* __restrict__ ssum) {
  int e = blockIdx.x * blockDim.x + threadIdx.x;
  if (e >= EP) return;
  int d = edge_dst(ei, e);
#pragma unroll
  for (int hd = 0; hd < Hh; ++hd) {
    float ex = __expf(ebuf[(size_t)e * Hh + hd] - m[d * Hh + hd]);
    ebuf[(size_t)e * Hh + hd] = ex;
    atomicAdd(&ssum[d * Hh + hd], ex);
  }
}

// out[dst] += h[src] * alpha ; one wave per edge, lanes stride channels
__global__ void edge_scatter_kernel(const int* __restrict__ ei,
                                    const float* __restrict__ h,
                                    const float* __restrict__ ebuf,
                                    const float* __restrict__ ssum,
                                    float* __restrict__ out) {
  int e = blockIdx.x * (blockDim.x >> 5) + (threadIdx.x >> 5);
  if (e >= EP) return;
  int lane = threadIdx.x & 31;
  int s = edge_src(ei, e), d = edge_dst(ei, e);
  const float* hs = h + (size_t)s * Dd;
  float* od = out + (size_t)d * Dd;
#pragma unroll
  for (int c = lane; c < Dd; c += 32) {
    int hd = c >> 6;
    float alpha = ebuf[(size_t)e * Hh + hd] / (ssum[d * Hh + hd] + 1e-16f);
    atomicAdd(&od[c], hs[c] * alpha);
  }
}

// v = elu(agg + b) ; write f32 (in-place ok) and f16 copy
__global__ void bias_elu_cvt_kernel(const float* __restrict__ agg,
                                    const float* __restrict__ b,
                                    float* __restrict__ out32,
                                    _Float16* __restrict__ out16,
                                    int rows, int cols) {
  size_t n = (size_t)rows * cols;
  size_t i = (size_t)blockIdx.x * blockDim.x + threadIdx.x;
  size_t st = (size_t)gridDim.x * blockDim.x;
  for (; i < n; i += st) {
    float v = agg[i] + b[i % cols];
    v = (v > 0.0f) ? v : (__expf(v) - 1.0f);
    out32[i] = v;
    if (out16) out16[i] = (_Float16)v;
  }
}

// ------------------------------- pooling -----------------------------------

__global__ void func_scatter_kernel(const int* __restrict__ fm,
                                    const int* __restrict__ gm,
                                    const float* __restrict__ h,
                                    float* __restrict__ func) {
  int node = blockIdx.x * (blockDim.x >> 5) + (threadIdx.x >> 5);
  if (node >= Nn) return;
  if (fm[node] < 0) return;
  int f = gm[node];
  int lane = threadIdx.x & 31;
  const float* hp = h + (size_t)node * Dd;
  float* fp = func + (size_t)f * Dd;
#pragma unroll
  for (int c = lane; c < Dd; c += 32) atomicAdd(&fp[c], hp[c]);
}

__global__ void pool_scatter_kernel(const int* __restrict__ seg,
                                    const float* __restrict__ v,
                                    float* __restrict__ sums,
                                    float* __restrict__ cnt, int rows) {
  int r = blockIdx.x * (blockDim.x >> 5) + (threadIdx.x >> 5);
  if (r >= rows) return;
  int g = seg[r];
  int lane = threadIdx.x & 31;
  const float* vp = v + (size_t)r * Dd;
  float* sp = sums + (size_t)g * Dd;
#pragma unroll
  for (int c = lane; c < Dd; c += 32) atomicAdd(&sp[c], vp[c]);
  if (lane == 0) atomicAdd(&cnt[g], 1.0f);
}

__global__ void pool_div_kernel(float* __restrict__ sums,
                                const float* __restrict__ cnt) {
  int i = blockIdx.x * blockDim.x + threadIdx.x;
  if (i >= Gg * Dd) return;
  sums[i] = sums[i] / fmaxf(cnt[i / Dd], 1.0f);
}

__global__ void concat_cvt_kernel(const float* __restrict__ hier,
                                  const float* __restrict__ flat,
                                  _Float16* __restrict__ cat16) {
  int i = blockIdx.x * blockDim.x + threadIdx.x;   // over Gg*2*Dd
  if (i >= Gg * 2 * Dd) return;
  int g = i / (2 * Dd), j = i % (2 * Dd);
  float v = (j < Dd) ? hier[(size_t)g * Dd + j] : flat[(size_t)g * Dd + j - Dd];
  cat16[i] = (_Float16)v;
}

__global__ void gate_combine_kernel(const float* __restrict__ logit,
                                    const float* __restrict__ hier,
                                    const float* __restrict__ flat,
                                    _Float16* __restrict__ hagg16) {
  int i = blockIdx.x * blockDim.x + threadIdx.x;   // over Gg*Dd
  if (i >= Gg * Dd) return;
  float gt = 1.0f / (1.0f + __expf(-logit[i]));
  hagg16[i] = (_Float16)(gt * hier[i] + (1.0f - gt) * flat[i]);
}

// L2-normalize each OUTd-wide row; block = 128 threads (4 waves) per row
__global__ void normalize_rows_kernel(const float* __restrict__ in,
                                      float* __restrict__ out) {
  int r = blockIdx.x;
  int t = threadIdx.x;
  float v = in[(size_t)r * OUTd + t];
  float ss = v * v;
#pragma unroll
  for (int o = 16; o > 0; o >>= 1) ss += __shfl_down(ss, o, 32);
  __shared__ float red[4];
  if ((t & 31) == 0) red[t >> 5] = ss;
  __syncthreads();
  float tot = red[0] + red[1] + red[2] + red[3];
  float nrm = fmaxf(sqrtf(tot), 1e-12f);
  out[(size_t)r * OUTd + t] = v / nrm;
}

// ------------------------------- driver ------------------------------------

extern "C" void kernel_launch(void* const* d_in, const int* in_sizes, int n_in,
                              void* d_out, int out_size, void* d_ws, size_t ws_size,
                              hipStream_t stream) {
  (void)in_sizes; (void)n_in; (void)out_size; (void)ws_size;

  const float* x    = (const float*)d_in[0];
  const int*   ei   = (const int*)d_in[1];
  const int*   fm   = (const int*)d_in[2];
  const int*   gm   = (const int*)d_in[3];
  const int*   n2g  = (const int*)d_in[4];
  const int*   f2g  = (const int*)d_in[5];
  const float* W[3]    = {(const float*)d_in[7],  (const float*)d_in[11], (const float*)d_in[15]};
  const float* asrc[3] = {(const float*)d_in[8],  (const float*)d_in[12], (const float*)d_in[16]};
  const float* adst[3] = {(const float*)d_in[9],  (const float*)d_in[13], (const float*)d_in[17]};
  const float* bb[3]   = {(const float*)d_in[10], (const float*)d_in[14], (const float*)d_in[18]};
  const float* Wf = (const float*)d_in[19]; const float* bf = (const float*)d_in[20];
  const float* Wg = (const float*)d_in[21]; const float* bg = (const float*)d_in[22];
  const float* Wo = (const float*)d_in[23]; const float* bo = (const float*)d_in[24];
  float* outp = (float*)d_out;

  // workspace carve-up
  char* ws = (char*)d_ws;
  size_t off = 0;
  auto alloc = [&](size_t bytes) -> void* {
    off = (off + 255) & ~(size_t)255;
    void* p = ws + off;
    off += bytes;
    return p;
  };
  _Float16* xh    = (_Float16*)alloc((size_t)Nn * FIN * 2);
  _Float16* w0t   = (_Float16*)alloc((size_t)Dd * FIN * 2);
  _Float16* w1t   = (_Float16*)alloc((size_t)Dd * Dd * 2);
  _Float16* w2t   = (_Float16*)alloc((size_t)Dd * Dd * 2);
  _Float16* wft   = (_Float16*)alloc((size_t)Dd * Dd * 2);
  _Float16* wgt   = (_Float16*)alloc((size_t)Dd * 2 * Dd * 2);
  _Float16* wot   = (_Float16*)alloc((size_t)OUTd * Dd * 2);
  float*    bufA  = (float*)alloc((size_t)Nn * Dd * 4);   // gemm out (pre-attn h)
  float*    bufB  = (float*)alloc((size_t)Nn * Dd * 4);   // agg / post-elu h
  _Float16* hin16 = (_Float16*)alloc((size_t)Nn * Dd * 2);
  float*    ssrc  = (float*)alloc((size_t)Nn * Hh * 4);
  float*    sdst  = (float*)alloc((size_t)Nn * Hh * 4);
  float*    mbuf  = (float*)alloc((size_t)Nn * Hh * 4);
  float*    ssum  = (float*)alloc((size_t)Nn * Hh * 4);
  float*    ebuf  = (float*)alloc((size_t)EP * Hh * 4);
  float*    func32  = (float*)alloc((size_t)NFn * Dd * 4);
  _Float16* func16  = (_Float16*)alloc((size_t)NFn * Dd * 2);
  float*    funcOut = (float*)alloc((size_t)NFn * Dd * 4);
  float*    hier    = (float*)alloc((size_t)Gg * Dd * 4);
  float*    flat    = (float*)alloc((size_t)Gg * Dd * 4);
  float*    cntH    = (float*)alloc((size_t)Gg * 4);
  float*    cntF    = (float*)alloc((size_t)Gg * 4);
  _Float16* cat16   = (_Float16*)alloc((size_t)Gg * 2 * Dd * 2);
  float*    glogit  = (float*)alloc((size_t)Gg * Dd * 4);
  _Float16* hagg16  = (_Float16*)alloc((size_t)Gg * Dd * 2);
  float*    outraw  = (float*)alloc((size_t)Gg * OUTd * 4);

  // ---- precision prep ----
  cvt_f32_f16_kernel<<<4096, 256, 0, stream>>>(x, xh, (size_t)Nn * FIN);
  transpose_w_kernel<<<512, 256, 0, stream>>>(W[0], w0t, FIN, Dd);
  transpose_w_kernel<<<512, 256, 0, stream>>>(W[1], w1t, Dd, Dd);
  transpose_w_kernel<<<512, 256, 0, stream>>>(W[2], w2t, Dd, Dd);
  transpose_w_kernel<<<512, 256, 0, stream>>>(Wf, wft, Dd, Dd);
  transpose_w_kernel<<<512, 256, 0, stream>>>(Wg, wgt, 2 * Dd, Dd);
  transpose_w_kernel<<<512, 256, 0, stream>>>(Wo, wot, Dd, OUTd);

  const int edgeBlocks  = (EP + 255) / 256;
  const int waveBlocksE = (EP + 7) / 8;
  const int waveBlocksN = (Nn + 7) / 8;

  // ---- 3 GAT layers ----
  const _Float16* inAct = xh;
  const _Float16* wt[3] = {w0t, w1t, w2t};
  const int kdim[3] = {FIN, Dd, Dd};
  for (int L = 0; L < 3; ++L) {
    // h = act_in @ W_L   (raw pre-attention features)
    dim3 gg(Nn / 16, (Dd + 255) / 256);
    gemm_wmma_kernel<<<gg, 128, 0, stream>>>(inAct, wt[L], nullptr, bufA,
                                             Nn, kdim[L], Dd, 0);
    // attention scores
    attn_scores_kernel<<<(Nn * Hh + 255) / 256, 256, 0, stream>>>(
        bufA, asrc[L], adst[L], ssrc, sdst);
    fill_f32_kernel<<<256, 256, 0, stream>>>(mbuf, -3.0e38f, (size_t)Nn * Hh);
    hipMemsetAsync(ssum, 0, (size_t)Nn * Hh * 4, stream);
    hipMemsetAsync(bufB, 0, (size_t)Nn * Dd * 4, stream);
    edge_logits_kernel<<<edgeBlocks, 256, 0, stream>>>(ei, ssrc, sdst, ebuf, mbuf);
    edge_exp_kernel<<<edgeBlocks, 256, 0, stream>>>(ei, mbuf, ebuf, ssum);
    edge_scatter_kernel<<<waveBlocksE, 256, 0, stream>>>(ei, bufA, ebuf, ssum, bufB);
    // h = elu(agg + b)  -> f32 (bufB in place) + f16 for next GEMM
    bias_elu_cvt_kernel<<<4096, 256, 0, stream>>>(bufB, bb[L], bufB, hin16, Nn, Dd);
    inAct = hin16;
  }
  // bufB now holds final node embeddings h (f32), hin16 the f16 copy.

  // ---- function-level pooling + MLP ----
  hipMemsetAsync(func32, 0, (size_t)NFn * Dd * 4, stream);
  func_scatter_kernel<<<waveBlocksN, 256, 0, stream>>>(fm, gm, bufB, func32);
  cvt_f32_f16_kernel<<<1024, 256, 0, stream>>>(func32, func16, (size_t)NFn * Dd);
  {
    dim3 gg(NFn / 16, (Dd + 255) / 256);
    gemm_wmma_kernel<<<gg, 128, 0, stream>>>(func16, wft, bf, funcOut,
                                             NFn, Dd, Dd, 1 /*elu*/);
  }

  // ---- mean pools ----
  hipMemsetAsync(hier, 0, (size_t)Gg * Dd * 4, stream);
  hipMemsetAsync(flat, 0, (size_t)Gg * Dd * 4, stream);
  hipMemsetAsync(cntH, 0, (size_t)Gg * 4, stream);
  hipMemsetAsync(cntF, 0, (size_t)Gg * 4, stream);
  pool_scatter_kernel<<<(NFn + 7) / 8, 256, 0, stream>>>(f2g, funcOut, hier, cntH, NFn);
  pool_scatter_kernel<<<waveBlocksN, 256, 0, stream>>>(n2g, bufB, flat, cntF, Nn);
  pool_div_kernel<<<(Gg * Dd + 255) / 256, 256, 0, stream>>>(hier, cntH);
  pool_div_kernel<<<(Gg * Dd + 255) / 256, 256, 0, stream>>>(flat, cntF);

  // ---- gate ----
  concat_cvt_kernel<<<(Gg * 2 * Dd + 255) / 256, 256, 0, stream>>>(hier, flat, cat16);
  {
    dim3 gg(Gg / 16, (Dd + 255) / 256);
    gemm_wmma_kernel<<<gg, 128, 0, stream>>>(cat16, wgt, bg, glogit,
                                             Gg, 2 * Dd, Dd, 0);
  }
  gate_combine_kernel<<<(Gg * Dd + 255) / 256, 256, 0, stream>>>(glogit, hier, flat, hagg16);

  // ---- output projection + row normalize ----
  {
    dim3 gg(Gg / 16, (OUTd + 255) / 256);
    gemm_wmma_kernel<<<gg, 128, 0, stream>>>(hagg16, wot, bo, outraw,
                                             Gg, Dd, OUTd, 0);
  }
  normalize_rows_kernel<<<Gg, OUTd, 0, stream>>>(outraw, outp);
}
